// SelfAttention_17205638988415
// MI455X (gfx1250) — compile-verified, matrix operand order checked by reading, not run on hardware
//
#include <hip/hip_runtime.h>
#include <hip/hip_bf16.h>

typedef __attribute__((ext_vector_type(16))) _Float16 v16h;
typedef __attribute__((ext_vector_type(8)))  float    v8f;
typedef __attribute__((ext_vector_type(4)))  unsigned int v4u;

#define B_   4
#define S_   2048
#define DIN  1024
#define HID_ 1024
#define H_   16
#define DH_  64
#define OUT_ 128
#define ROWS (B_ * S_)          // 8192

union Frag { v16h h; v4u u[2]; };

// NT-layout fragment load: matrix stored row-major with the WMMA K dimension
// contiguous (stride `ld` halves). Matches the documented 16-bit A layout:
// lanes 0-15 hold rows 0-15 with K = {ks..ks+7, ks+16..ks+23}, ks = 8*(lane/16).
__device__ __forceinline__ v16h frag_nt(const _Float16* p, int ld) {
    const int lane = threadIdx.x & 31;
    const _Float16* r = p + (lane & 15) * ld + ((lane >> 4) << 3);
    Frag f;
    f.u[0] = *(const v4u*)(r);
    f.u[1] = *(const v4u*)(r + 16);
    return f.h;
}

__device__ __forceinline__ v8f wmma_f16(v16h a, v16h b, v8f c) {
    return __builtin_amdgcn_wmma_f32_16x16x32_f16(
        /*neg_a=*/false, a, /*neg_b=*/false, b,
        /*c_mod=*/(short)0, c, /*reuse_a=*/false, /*reuse_b=*/false);
}

// Async global -> LDS copy, 16 bytes per lane (CDNA5 ASYNCcnt path).
// Generic-pointer low 32 bits are the LDS byte offset (aperture rule).
__device__ __forceinline__ void async_b128(const void* gsrc, void* ldst) {
    unsigned           l = (unsigned)(unsigned long long)(uintptr_t)ldst;
    unsigned long long g = (unsigned long long)(uintptr_t)gsrc;
    asm volatile("global_load_async_to_lds_b128 %0, %1, off"
                 :: "v"(l), "v"(g) : "memory");
}
__device__ __forceinline__ void wait_async() {
    asm volatile("s_wait_asynccnt 0x0" ::: "memory");
}

// ---------------------------------------------------------------- weights cvt
__global__ void cvt_f32_f16(const float* __restrict__ src,
                            _Float16* __restrict__ dst, int n) {
    int i = blockIdx.x * blockDim.x + threadIdx.x;
    if (i < n) dst[i] = (_Float16)src[i];
}

// ---------------------------------------------------------------- QKV GEMM
// Y[r,o] = sum_k X[r,k] * W[o,k] + bias[o]  -> stored as (B,H,S,DH) f16
__global__ __launch_bounds__(256, 1) void qkv_gemm(
    const float*    __restrict__ X,     // ROWS x DIN  (f32)
    const _Float16* __restrict__ W,     // HID  x DIN  (f16)
    const float*    __restrict__ bias,  // HID
    _Float16*       __restrict__ dst)   // (B,H,S,DH)
{
    __shared__ __align__(16) _Float16 As[2][128][40];
    __shared__ __align__(16) _Float16 Bs[2][128][40];

    const int tid  = threadIdx.x;
    const int w    = tid >> 5;
    const int lane = tid & 31;
    const int m0   = blockIdx.x * 128;
    const int n0   = blockIdx.y * 128;

    const v8f zc = {};
    v8f acc[8];
    #pragma unroll
    for (int t = 0; t < 8; ++t) acc[t] = zc;

    const int ar = tid >> 1;          // staging row 0..127
    const int ak = (tid & 1) << 4;    // staging k-offset 0/16

    auto stage = [&](int buf, int k0) {
        // A: 16 f32 -> f16 per thread (conversion must go through VGPRs)
        const float* srcA = X + (size_t)(m0 + ar) * DIN + k0 + ak;
        #pragma unroll
        for (int i = 0; i < 16; ++i) As[buf][ar][ak + i] = (_Float16)srcA[i];
        // B: pure f16 copy -> async global->LDS, bypasses VGPRs
        const _Float16* srcB = W + (size_t)(n0 + ar) * DIN + k0 + ak;
        async_b128(srcB,     &Bs[buf][ar][ak]);
        async_b128(srcB + 8, &Bs[buf][ar][ak + 8]);
    };

    stage(0, 0);
    wait_async();
    __syncthreads();

    for (int k0 = 0; k0 < DIN; k0 += 32) {
        const int buf = (k0 >> 5) & 1;
        if (k0 + 32 < DIN) {
            stage(buf ^ 1, k0 + 32);
            __builtin_prefetch(X + (size_t)(m0 + ar) * DIN + k0 + 64, 0, 1);
        }

        // distinct registers for all fragments -> ds loads overlap the WMMA burst
        const v16h a = frag_nt(&As[buf][w * 16][0], 40);
        v16h bf[8];
        #pragma unroll
        for (int t = 0; t < 8; ++t) bf[t] = frag_nt(&Bs[buf][t * 16][0], 40);
        #pragma unroll
        for (int t = 0; t < 8; ++t) acc[t] = wmma_f16(a, bf[t], acc[t]);

        wait_async();
        __syncthreads();
    }

    // epilogue: bias add, f16 convert, head-interleaved store
    const int mrow = m0 + w * 16 + ((lane >> 4) << 3);
    #pragma unroll
    for (int t = 0; t < 8; ++t) {
        const int o  = n0 + t * 16 + (lane & 15);
        const int h  = o >> 6;       // / DH_
        const int dh = o & 63;
        const float bv = bias[o];
        #pragma unroll
        for (int r = 0; r < 8; ++r) {
            const int row = mrow + r;
            const int bb  = row >> 11;       // / S_
            const int ss  = row & 2047;
            const float v = acc[t][r] + bv;
            dst[(((size_t)(bb * H_ + h)) * S_ + ss) * DH_ + dh] = (_Float16)v;
        }
    }
}

// ---------------------------------------------------------------- attention
__global__ __launch_bounds__(256, 1) void attention(
    const _Float16* __restrict__ Q16,   // (B,H,S,DH)
    const _Float16* __restrict__ K16,
    const _Float16* __restrict__ V16,
    const int*      __restrict__ mask,  // (B,1,1,S)
    _Float16*       __restrict__ O16)   // ROWS x HID (row = b*S+s, col = h*DH+dh)
{
    __shared__ __align__(16) _Float16 Ks[2][32][72];     // keys x dh
    __shared__ __align__(16) _Float16 Vt[2][64][40];     // dh x keys (transposed)
    __shared__ __align__(16) _Float16 Ps[8][16][32];     // per-wave P tile

    const int tid  = threadIdx.x;
    const int w    = tid >> 5;
    const int lane = tid & 31;
    const int bh   = blockIdx.y;
    const int b    = bh >> 4;          // / H_
    const int h    = bh & 15;
    const int q0   = blockIdx.x * 128 + w * 16;

    const _Float16* Qb = Q16 + (size_t)bh * S_ * DH_;
    const _Float16* Kb = K16 + (size_t)bh * S_ * DH_;
    const _Float16* Vb = V16 + (size_t)bh * S_ * DH_;
    const int*      mb = mask + b * S_;

    // Q fragment (16 q-rows x 64 dh), split into two K=32 fragments
    const v16h qa0 = frag_nt(Qb + (size_t)q0 * DH_,      DH_);
    const v16h qa1 = frag_nt(Qb + (size_t)q0 * DH_ + 32, DH_);

    const v8f zc = {};
    v8f oacc[4];
    #pragma unroll
    for (int t = 0; t < 4; ++t) oacc[t] = zc;
    float mrow[8], lrow[8];
    #pragma unroll
    for (int r = 0; r < 8; ++r) { mrow[r] = -1e30f; lrow[r] = 0.0f; }

    auto stageKV = [&](int buf, int k0) {
        // K tile: pure f16 copy -> async
        const int kr = tid >> 3;
        const int kc = (tid & 7) << 3;
        async_b128(Kb + (size_t)(k0 + kr) * DH_ + kc, &Ks[buf][kr][kc]);
        // V tile transposed: Vt[dh][k] (data transform -> through VGPRs)
        #pragma unroll
        for (int i = 0; i < 8; ++i) {
            const int flat = tid * 8 + i;
            const int dh = flat >> 5, kk = flat & 31;
            Vt[buf][dh][kk] = Vb[(size_t)(k0 + kk) * DH_ + dh];
        }
    };

    stageKV(0, 0);
    wait_async();
    __syncthreads();

    for (int k0 = 0; k0 < S_; k0 += 32) {
        const int buf = (k0 >> 5) & 1;
        if (k0 + 32 < S_) stageKV(buf ^ 1, k0 + 32);

        // energy: two 16x16 tiles covering 32 keys, dot-dim = 64
        const v16h kf0a = frag_nt(&Ks[buf][0][0],   72);
        const v16h kf0b = frag_nt(&Ks[buf][0][32],  72);
        const v16h kf1a = frag_nt(&Ks[buf][16][0],  72);
        const v16h kf1b = frag_nt(&Ks[buf][16][32], 72);
        v8f s0 = zc, s1 = zc;
        s0 = wmma_f16(qa0, kf0a, s0);
        s0 = wmma_f16(qa1, kf0b, s0);
        s1 = wmma_f16(qa0, kf1a, s1);
        s1 = wmma_f16(qa1, kf1b, s1);

        const bool z0 = (mb[k0 +      (lane & 15)] == 0);
        const bool z1 = (mb[k0 + 16 + (lane & 15)] == 0);

        #pragma unroll
        for (int r = 0; r < 8; ++r) {
            const float e0 = z0 ? -1e10f : s0[r] * 0.125f;   // 1/sqrt(64)
            const float e1 = z1 ? -1e10f : s1[r] * 0.125f;
            // row max across the 16-lane group (wave32 halves)
            float mx = fmaxf(e0, e1);
            #pragma unroll
            for (int off = 8; off >= 1; off >>= 1)
                mx = fmaxf(mx, __shfl_xor(mx, off, 32));
            const float mnew  = fmaxf(mrow[r], mx);
            const float alpha = __expf(mrow[r] - mnew);
            const float p0 = __expf(e0 - mnew);
            const float p1 = __expf(e1 - mnew);
            float ps = p0 + p1;
            #pragma unroll
            for (int off = 8; off >= 1; off >>= 1)
                ps += __shfl_xor(ps, off, 32);
            lrow[r] = lrow[r] * alpha + ps;
            mrow[r] = mnew;
            #pragma unroll
            for (int t = 0; t < 4; ++t) oacc[t][r] *= alpha;
            // spill P (C layout) to LDS so we can re-fragment it as an A operand
            const int m = r + ((lane >> 4) << 3);
            Ps[w][m][lane & 15]        = (_Float16)p0;
            Ps[w][m][16 + (lane & 15)] = (_Float16)p1;
        }

        // O += P (16x32) x V (32x64)
        const v16h pa = frag_nt(&Ps[w][0][0], 32);
        v16h vf[4];
        #pragma unroll
        for (int t = 0; t < 4; ++t) vf[t] = frag_nt(&Vt[buf][t * 16][0], 40);
        #pragma unroll
        for (int t = 0; t < 4; ++t) oacc[t] = wmma_f16(pa, vf[t], oacc[t]);

        wait_async();
        __syncthreads();
    }

    // normalize and store O as f16, row = b*S+s, col = h*DH+dh
    const int srow_base = q0 + ((lane >> 4) << 3);
    #pragma unroll
    for (int t = 0; t < 4; ++t) {
        const int dh = t * 16 + (lane & 15);
        #pragma unroll
        for (int r = 0; r < 8; ++r) {
            const int srow = srow_base + r;
            const float v = oacc[t][r] / lrow[r];
            O16[((size_t)(b * S_ + srow)) * HID_ + h * DH_ + dh] = (_Float16)v;
        }
    }
}

// ---------------------------------------------------------------- FC + softmax(dim=1)
__global__ __launch_bounds__(256, 1) void fc_softmax(
    const _Float16* __restrict__ Xin,   // ROWS x HID (f16)
    const _Float16* __restrict__ Wfc,   // OUT x HID  (f16)
    const float*    __restrict__ bias,  // OUT
    float*          __restrict__ out)   // ROWS x OUT (f32)
{
    __shared__ __align__(16) _Float16 As[2][128][40];
    __shared__ __align__(16) _Float16 Bs[2][128][40];

    const int tid  = threadIdx.x;
    const int w    = tid >> 5;
    const int lane = tid & 31;
    const int m0   = blockIdx.x * 128;

    const v8f zc = {};
    v8f acc[8];
    #pragma unroll
    for (int t = 0; t < 8; ++t) acc[t] = zc;

    const int ar = tid >> 1;
    const int ak = (tid & 1) << 4;

    auto stage = [&](int buf, int k0) {
        const _Float16* sa = Xin + (size_t)(m0 + ar) * HID_ + k0 + ak;
        async_b128(sa,     &As[buf][ar][ak]);
        async_b128(sa + 8, &As[buf][ar][ak + 8]);
        const _Float16* sb = Wfc + (size_t)ar * HID_ + k0 + ak;
        async_b128(sb,     &Bs[buf][ar][ak]);
        async_b128(sb + 8, &Bs[buf][ar][ak + 8]);
    };

    stage(0, 0);
    wait_async();
    __syncthreads();

    for (int k0 = 0; k0 < HID_; k0 += 32) {
        const int buf = (k0 >> 5) & 1;
        if (k0 + 32 < HID_) stage(buf ^ 1, k0 + 32);

        const v16h a = frag_nt(&As[buf][w * 16][0], 40);
        v16h bf[8];
        #pragma unroll
        for (int t = 0; t < 8; ++t) bf[t] = frag_nt(&Bs[buf][t * 16][0], 40);
        #pragma unroll
        for (int t = 0; t < 8; ++t) acc[t] = wmma_f16(a, bf[t], acc[t]);

        wait_async();
        __syncthreads();
    }

    // softmax over reshape(-1,4,32) dim=1: columns {j, j+32, j+64, j+96}
    // => accumulator tiles {0,2,4,6} (cols j = lane&15) and {1,3,5,7} (j+16):
    // entirely per-lane, no cross-lane traffic.
    float bv[8];
    #pragma unroll
    for (int t = 0; t < 8; ++t) bv[t] = bias[t * 16 + (lane & 15)];
    const int rowb = m0 + w * 16 + ((lane >> 4) << 3);
    const int c = lane & 15;

    #pragma unroll
    for (int r = 0; r < 8; ++r) {
        float* orow = out + (size_t)(rowb + r) * OUT_;
        {   // even group -> cols c, c+32, c+64, c+96
            const float g0 = acc[0][r] + bv[0], g1 = acc[2][r] + bv[2];
            const float g2 = acc[4][r] + bv[4], g3 = acc[6][r] + bv[6];
            const float mx = fmaxf(fmaxf(g0, g1), fmaxf(g2, g3));
            const float p0 = __expf(g0 - mx), p1 = __expf(g1 - mx);
            const float p2 = __expf(g2 - mx), p3 = __expf(g3 - mx);
            const float inv = 1.0f / (p0 + p1 + p2 + p3);
            orow[c]      = p0 * inv; orow[c + 32] = p1 * inv;
            orow[c + 64] = p2 * inv; orow[c + 96] = p3 * inv;
        }
        {   // odd group -> cols c+16, c+48, c+80, c+112
            const float g0 = acc[1][r] + bv[1], g1 = acc[3][r] + bv[3];
            const float g2 = acc[5][r] + bv[5], g3 = acc[7][r] + bv[7];
            const float mx = fmaxf(fmaxf(g0, g1), fmaxf(g2, g3));
            const float p0 = __expf(g0 - mx), p1 = __expf(g1 - mx);
            const float p2 = __expf(g2 - mx), p3 = __expf(g3 - mx);
            const float inv = 1.0f / (p0 + p1 + p2 + p3);
            orow[c + 16] = p0 * inv; orow[c + 48] = p1 * inv;
            orow[c + 80] = p2 * inv; orow[c + 112] = p3 * inv;
        }
    }
}

// ---------------------------------------------------------------- launch
extern "C" void kernel_launch(void* const* d_in, const int* in_sizes, int n_in,
                              void* d_out, int out_size, void* d_ws, size_t ws_size,
                              hipStream_t stream) {
    (void)in_sizes; (void)n_in; (void)out_size; (void)ws_size;

    const float* query = (const float*)d_in[0];
    const float* key   = (const float*)d_in[1];
    const float* value = (const float*)d_in[2];
    const int*   mask  = (const int*)d_in[3];
    const float* wq_w  = (const float*)d_in[4];
    const float* wq_b  = (const float*)d_in[5];
    const float* wk_w  = (const float*)d_in[6];
    const float* wk_b  = (const float*)d_in[7];
    const float* wv_w  = (const float*)d_in[8];
    const float* wv_b  = (const float*)d_in[9];
    const float* fc_w  = (const float*)d_in[10];
    const float* fc_b  = (const float*)d_in[11];

    _Float16* ws    = (_Float16*)d_ws;
    _Float16* wq16  = ws;
    _Float16* wk16  = wq16 + (size_t)HID_ * DIN;
    _Float16* wv16  = wk16 + (size_t)HID_ * DIN;
    _Float16* wfc16 = wv16 + (size_t)HID_ * DIN;
    _Float16* q16   = wfc16 + (size_t)OUT_ * HID_;
    _Float16* k16   = q16 + (size_t)ROWS * HID_;
    _Float16* v16   = k16 + (size_t)ROWS * HID_;
    _Float16* o16   = v16 + (size_t)ROWS * HID_;

    const int nw = HID_ * DIN;
    cvt_f32_f16<<<nw / 256, 256, 0, stream>>>(wq_w, wq16, nw);
    cvt_f32_f16<<<nw / 256, 256, 0, stream>>>(wk_w, wk16, nw);
    cvt_f32_f16<<<nw / 256, 256, 0, stream>>>(wv_w, wv16, nw);
    cvt_f32_f16<<<(OUT_ * HID_) / 256, 256, 0, stream>>>(fc_w, wfc16, OUT_ * HID_);

    dim3 g1(ROWS / 128, HID_ / 128);
    qkv_gemm<<<g1, 256, 0, stream>>>(query, wq16, wq_b, q16);
    qkv_gemm<<<g1, 256, 0, stream>>>(key,   wk16, wk_b, k16);
    qkv_gemm<<<g1, 256, 0, stream>>>(value, wv16, wv_b, v16);

    attention<<<dim3(S_ / 128, B_ * H_), 256, 0, stream>>>(q16, k16, v16, mask, o16);

    fc_softmax<<<ROWS / 128, 256, 0, stream>>>(o16, wfc16, fc_b, (float*)d_out);
}
